// _FakeVllmLlamaAttention_22771916603759
// MI455X (gfx1250) — matrix operand order repaired and checked by reading, not compiled, not run on hardware
//
#include <hip/hip_runtime.h>
#include <hip/hip_bf16.h>

typedef __bf16 bf16_t;
typedef __attribute__((ext_vector_type(16))) __bf16 v16bf;
typedef __attribute__((ext_vector_type(8)))  float  v8f;
typedef __attribute__((ext_vector_type(4)))  unsigned int v4u;

#define WMMA_BF16(a, b, c) \
  __builtin_amdgcn_wmma_f32_16x16x32_bf16(false, (a), false, (b), (short)0, (c), false, false)

// ---- helpers -------------------------------------------------------------

__device__ __forceinline__ v16bf load16(const bf16_t* p) {
  union { v16bf v; v4u u[2]; } r;
  r.u[0] = *(const v4u*)(p);
  r.u[1] = *(const v4u*)(p + 8);
  return r.v;
}

// A-operand fragment: elems 0..15 = row[2*k0..2*k0+15]
__device__ __forceinline__ v16bf loadA16(const bf16_t* row, int k0) {
  union { v16bf v; v4u u[2]; } r;
  r.u[0] = *(const v4u*)(row + 2 * k0);
  r.u[1] = *(const v4u*)(row + 2 * k0 + 8);
  return r.v;
}

__device__ __forceinline__ v8f zero8() {
  v8f z;
#pragma unroll
  for (int i = 0; i < 8; ++i) z[i] = 0.0f;
  return z;
}

// ---- fp32 -> bf16 conversion --------------------------------------------

__global__ __launch_bounds__(256) void cvt_f32_to_bf16(const float* __restrict__ in,
                                                       bf16_t* __restrict__ out, int n8) {
  int i = blockIdx.x * 256 + threadIdx.x;
  if (i >= n8) return;
  const float4* p = (const float4*)in + (size_t)i * 2;
  float4 a = p[0], b = p[1];
  union { bf16_t h[8]; v4u u; } r;
  r.h[0] = (bf16_t)a.x; r.h[1] = (bf16_t)a.y; r.h[2] = (bf16_t)a.z; r.h[3] = (bf16_t)a.w;
  r.h[4] = (bf16_t)b.x; r.h[5] = (bf16_t)b.y; r.h[6] = (bf16_t)b.z; r.h[7] = (bf16_t)b.w;
  *((v4u*)out + i) = r.u;
}

// ---- tiled bf16 WMMA GEMM: C[M,N] = A[M,K] @ B[N,K]^T --------------------
// Block tile 128x256, wave tile 64x64 (4x4 WMMA). Double-buffered LDS with
// register-staged global loads.
// MODE 0: write bf16, scale columns < qcols by qscale.  MODE 1: write fp32.

template <int MODE>
__global__ __launch_bounds__(256) void gemm_bf16_k(const bf16_t* __restrict__ A,
                                                   const bf16_t* __restrict__ B,
                                                   void* __restrict__ Cout,
                                                   int N, int K, float qscale, int qcols) {
  __shared__ bf16_t As[2][128][40];  // 128 rows x 32 k (+8 pad)
  __shared__ bf16_t Bs[2][256][40];  // 256 rows x 32 k (+8 pad)

  const int tid  = threadIdx.x;
  const int lane = tid & 31;
  const int wid  = tid >> 5;
  const int waveM = wid & 1;      // 2 waves in M (64 rows each)
  const int waveN = wid >> 1;     // 4 waves in N (64 cols each)
  const int bM = blockIdx.y * 128, bN = blockIdx.x * 256;
  const int arow = tid >> 1, ahalf = (tid & 1) * 16;
  const int m15 = lane & 15, khi = lane >> 4;

  v8f acc[4][4];
#pragma unroll
  for (int mt = 0; mt < 4; ++mt)
#pragma unroll
    for (int nt = 0; nt < 4; ++nt) acc[mt][nt] = zero8();

  const bf16_t* ga = A + (size_t)(bM + arow) * K + ahalf;  // 16 bf16 per thread
  const bf16_t* gb = B + (size_t)(bN + tid) * K;           // 32 bf16 per thread
  const int nkt = K >> 5;

  // prologue: tile 0 -> regs -> LDS buf 0
  v4u ra0 = *(const v4u*)ga, ra1 = *(const v4u*)(ga + 8);
  v4u rb0 = *(const v4u*)gb,        rb1 = *(const v4u*)(gb + 8);
  v4u rb2 = *(const v4u*)(gb + 16), rb3 = *(const v4u*)(gb + 24);
  *(v4u*)&As[0][arow][ahalf]     = ra0;
  *(v4u*)&As[0][arow][ahalf + 8] = ra1;
  *(v4u*)&Bs[0][tid][0]  = rb0;
  *(v4u*)&Bs[0][tid][8]  = rb1;
  *(v4u*)&Bs[0][tid][16] = rb2;
  *(v4u*)&Bs[0][tid][24] = rb3;

  for (int t = 0; t < nkt; ++t) {
    const int cur = t & 1, nxt = cur ^ 1;
    __syncthreads();

    // issue global loads for tile t+1 (hidden behind the WMMA burst)
    if (t + 1 < nkt) {
      const bf16_t* ga2 = ga + (size_t)(t + 1) * 32;
      const bf16_t* gb2 = gb + (size_t)(t + 1) * 32;
      ra0 = *(const v4u*)ga2;        ra1 = *(const v4u*)(ga2 + 8);
      rb0 = *(const v4u*)gb2;        rb1 = *(const v4u*)(gb2 + 8);
      rb2 = *(const v4u*)(gb2 + 16); rb3 = *(const v4u*)(gb2 + 24);
    }

    // compute tile t from LDS[cur]
    v16bf a[4];
#pragma unroll
    for (int mt = 0; mt < 4; ++mt)
      a[mt] = loadA16(&As[cur][waveM * 64 + mt * 16 + m15][0], khi * 8);
#pragma unroll
    for (int nt = 0; nt < 4; ++nt) {
      v16bf b = load16(&Bs[cur][waveN * 64 + nt * 16 + m15][khi * 16]);
#pragma unroll
      for (int mt = 0; mt < 4; ++mt)
        acc[mt][nt] = WMMA_BF16(a[mt], b, acc[mt][nt]);
    }

    // stage tile t+1 into the other LDS buffer (barrier at top of t+1 orders it)
    if (t + 1 < nkt) {
      *(v4u*)&As[nxt][arow][ahalf]     = ra0;
      *(v4u*)&As[nxt][arow][ahalf + 8] = ra1;
      *(v4u*)&Bs[nxt][tid][0]  = rb0;
      *(v4u*)&Bs[nxt][tid][8]  = rb1;
      *(v4u*)&Bs[nxt][tid][16] = rb2;
      *(v4u*)&Bs[nxt][tid][24] = rb3;
    }
  }

#pragma unroll
  for (int mt = 0; mt < 4; ++mt)
#pragma unroll
    for (int nt = 0; nt < 4; ++nt)
#pragma unroll
      for (int i = 0; i < 8; ++i) {
        int row = bM + waveM * 64 + mt * 16 + i + khi * 8;
        int col = bN + waveN * 64 + nt * 16 + m15;
        float v = acc[mt][nt][i];
        if (MODE == 0) {
          if (col < qcols) v *= qscale;
          ((bf16_t*)Cout)[(size_t)row * N + col] = (bf16_t)v;
        } else {
          ((float*)Cout)[(size_t)row * N + col] = v;
        }
      }
}

// ---- causal GQA flash attention -----------------------------------------
// Grid: (T/32, N_KV_HEADS). 8 waves = 4 q-heads x 2 row-subtiles(16).
// K/V tiles double-buffered with register-staged global loads.
// Row sums of P come from an extra WMMA against an all-ones B operand
// (no cross-lane sum reductions). P is staged packed (2 bf16 per ds_store_b32)
// using a key permutation applied consistently to P columns and V rows.

__global__ __launch_bounds__(256) void attn_k(const bf16_t* __restrict__ qkv,
                                              bf16_t* __restrict__ ctx) {
  __shared__ bf16_t Kt[2][32][136];   // 32 keys x 128 dims (+8 pad)
  __shared__ bf16_t Vt[2][128][40];   // transposed: 128 dims x 32 permuted keys (+8 pad)
  __shared__ bf16_t Pst[8][16][40];   // per-wave P staging (permuted key order)

  const int tid  = threadIdx.x;
  const int lane = tid & 31;
  const int wid  = tid >> 5;
  const int qb   = blockIdx.x;
  const int kvh  = blockIdx.y;
  const int g    = wid >> 1;
  const int sub  = wid & 1;
  const int qhead = kvh * 4 + g;
  const int qr0   = qb * 32 + sub * 16;
  const int m15 = lane & 15, khi = lane >> 4;

  // Q fragments (A-operand), 4 chunks of 32 head-dims
  v16bf aq[4];
  {
    const bf16_t* qrow = qkv + (size_t)(qr0 + m15) * 6144 + qhead * 128;
#pragma unroll
    for (int c = 0; c < 4; ++c) aq[c] = loadA16(qrow + c * 32, khi * 8);
  }

  // all-ones B operand (32x16 ones matrix) for row-sum WMMA
  v16bf ones;
#pragma unroll
  for (int e = 0; e < 16; ++e) ones[e] = (bf16_t)1.0f;

  v8f o[8];
#pragma unroll
  for (int dt = 0; dt < 8; ++dt) o[dt] = zero8();
  v8f osum = zero8();
  float mrow[8];
#pragma unroll
  for (int i = 0; i < 8; ++i) mrow[i] = -3.0e38f;

  const int key = tid >> 3, ch = tid & 7;
  // permuted key position: 2j <- key j, 2j+1 <- key 16+j
  const int pkey = (key < 16) ? (2 * key) : (2 * (key - 16) + 1);
  const int nkb = qb + 1;

  // register staging for K/V tiles
  v4u rk0, rk1;
  union { v4u u[2]; bf16_t h[16]; } rv;

  // prologue: key block 0 -> regs -> LDS buf 0
  {
    const bf16_t* kp = qkv + (size_t)key * 6144 + 4096 + kvh * 128 + ch * 16;
    rk0 = *(const v4u*)kp; rk1 = *(const v4u*)(kp + 8);
    const bf16_t* vp = qkv + (size_t)key * 6144 + 5120 + kvh * 128 + ch * 16;
    rv.u[0] = *(const v4u*)vp; rv.u[1] = *(const v4u*)(vp + 8);
    *(v4u*)&Kt[0][key][ch * 16]     = rk0;
    *(v4u*)&Kt[0][key][ch * 16 + 8] = rk1;
#pragma unroll
    for (int j = 0; j < 16; ++j) Vt[0][ch * 16 + j][pkey] = rv.h[j];
  }

  for (int kb = 0; kb < nkb; ++kb) {
    const int cur = kb & 1, nxt = cur ^ 1;
    __syncthreads();

    // issue global loads for key block kb+1
    if (kb + 1 < nkb) {
      const bf16_t* kp = qkv + (size_t)((kb + 1) * 32 + key) * 6144 + 4096 + kvh * 128 + ch * 16;
      rk0 = *(const v4u*)kp; rk1 = *(const v4u*)(kp + 8);
      const bf16_t* vp = qkv + (size_t)((kb + 1) * 32 + key) * 6144 + 5120 + kvh * 128 + ch * 16;
      rv.u[0] = *(const v4u*)vp; rv.u[1] = *(const v4u*)(vp + 8);
    }

    // S = Q . K^T (two 16-key column tiles, original key order)
    v8f s0 = zero8(), s1 = zero8();
#pragma unroll
    for (int c = 0; c < 4; ++c) {
      v16bf b0 = load16(&Kt[cur][m15][c * 32 + khi * 16]);
      v16bf b1 = load16(&Kt[cur][16 + m15][c * 32 + khi * 16]);
      s0 = WMMA_BF16(aq[c], b0, s0);
      s1 = WMMA_BF16(aq[c], b1, s1);
    }

    // online softmax (causal mask; only the MAX needs cross-lane reduction)
    const int col0 = kb * 32 + m15, col1 = col0 + 16;
    float alpha[8];
#pragma unroll
    for (int i = 0; i < 8; ++i) {
      int row = qr0 + khi * 8 + i;
      float v0 = (col0 <= row) ? s0[i] : -3.0e38f;
      float v1 = (col1 <= row) ? s1[i] : -3.0e38f;
      float mx = fmaxf(v0, v1);
#pragma unroll
      for (int d = 1; d < 16; d <<= 1) mx = fmaxf(mx, __shfl_xor(mx, d, 32));
      float mnew = fmaxf(mrow[i], mx);
      alpha[i] = __expf(mrow[i] - mnew);
      mrow[i]  = mnew;
      s0[i] = __expf(v0 - mnew);
      s1[i] = __expf(v1 - mnew);
    }
#pragma unroll
    for (int dt = 0; dt < 8; ++dt)
#pragma unroll
      for (int i = 0; i < 8; ++i) o[dt][i] *= alpha[i];
#pragma unroll
    for (int i = 0; i < 8; ++i) osum[i] *= alpha[i];

    // C-layout P -> A-layout via per-wave LDS staging (packed, permuted keys)
#pragma unroll
    for (int i = 0; i < 8; ++i) {
      union { unsigned int u; bf16_t h[2]; } pk;
      pk.h[0] = (bf16_t)s0[i];
      pk.h[1] = (bf16_t)s1[i];
      *(unsigned int*)&Pst[wid][khi * 8 + i][2 * m15] = pk.u;
    }
    v16bf ap = loadA16(&Pst[wid][m15][0], khi * 8);

    // O += P . V (8 dim-tiles of 16), row sums += P . 1
#pragma unroll
    for (int h = 0; h < 2; ++h) {
      v16bf bv[4];
#pragma unroll
      for (int q = 0; q < 4; ++q)
        bv[q] = load16(&Vt[cur][(h * 4 + q) * 16 + m15][khi * 16]);
#pragma unroll
      for (int q = 0; q < 4; ++q)
        o[h * 4 + q] = WMMA_BF16(ap, bv[q], o[h * 4 + q]);
    }
    osum = WMMA_BF16(ap, ones, osum);

    // stage key block kb+1 into the other buffer
    if (kb + 1 < nkb) {
      *(v4u*)&Kt[nxt][key][ch * 16]     = rk0;
      *(v4u*)&Kt[nxt][key][ch * 16 + 8] = rk1;
#pragma unroll
      for (int j = 0; j < 16; ++j) Vt[nxt][ch * 16 + j][pkey] = rv.h[j];
    }
  }

  // finalize and write ctx (bf16): osum holds per-row softmax denominators
  float inv[8];
#pragma unroll
  for (int i = 0; i < 8; ++i) inv[i] = 1.0f / osum[i];
#pragma unroll
  for (int dt = 0; dt < 8; ++dt)
#pragma unroll
    for (int i = 0; i < 8; ++i) {
      int row = qr0 + khi * 8 + i;
      float v = o[dt][i] * inv[i];
      ctx[(size_t)row * 4096 + qhead * 128 + dt * 16 + m15] = (bf16_t)v;
    }
}

// ---- launch --------------------------------------------------------------

extern "C" void kernel_launch(void* const* d_in, const int* in_sizes, int n_in,
                              void* d_out, int out_size, void* d_ws, size_t ws_size,
                              hipStream_t stream) {
  // inputs: positions(int)[2048], hidden[2048*4096] f32, w_qkv[6144*4096] f32, w_o[4096*4096] f32
  const float* hidden = (const float*)d_in[1];
  const float* w_qkv  = (const float*)d_in[2];
  const float* w_o    = (const float*)d_in[3];
  float* out = (float*)d_out;

  bf16_t* hs   = (bf16_t*)d_ws;                   // [2048,4096]
  bf16_t* wqkv = hs   + (size_t)2048 * 4096;      // [6144,4096]
  bf16_t* wo   = wqkv + (size_t)6144 * 4096;      // [4096,4096]
  bf16_t* qkv  = wo   + (size_t)4096 * 4096;      // [2048,6144]
  bf16_t* ctx  = qkv  + (size_t)2048 * 6144;      // [2048,4096]

  const int n_hs   = 2048 * 4096 / 8;
  const int n_wqkv = 6144 * 4096 / 8;
  const int n_wo   = 4096 * 4096 / 8;
  cvt_f32_to_bf16<<<n_hs   / 256, 256, 0, stream>>>(hidden, hs,   n_hs);
  cvt_f32_to_bf16<<<n_wqkv / 256, 256, 0, stream>>>(w_qkv,  wqkv, n_wqkv);
  cvt_f32_to_bf16<<<n_wo   / 256, 256, 0, stream>>>(w_o,    wo,   n_wo);

  // QKV projection: [2048,4096] x [6144,4096]^T -> bf16 [2048,6144], Q pre-scaled
  gemm_bf16_k<0><<<dim3(24, 16), 256, 0, stream>>>(hs, wqkv, qkv, 6144, 4096,
                                                   0.08838834764831845f, 4096);
  // causal GQA flash attention -> bf16 ctx [2048,4096]
  attn_k<<<dim3(64, 8), 256, 0, stream>>>(qkv, ctx);
  // output projection: [2048,4096] x [4096,4096]^T -> fp32 d_out
  gemm_bf16_k<1><<<dim3(16, 16), 256, 0, stream>>>(ctx, wo, out, 4096, 4096, 1.0f, 0);
}
